// LinearAttentionARMA_57389353009770
// MI455X (gfx1250) — compile-verified, matrix operand order checked by reading, not run on hardware
//
#include <hip/hip_runtime.h>
#include <hip/hip_bf16.h>
#include <math.h>

// Problem geometry (fixed by the reference)
#define BB 2
#define LL 1024
#define DD 1024
#define HH 16
#define DH 64
#define MM (BB * LL)   // 2048 tokens
#define CHK 64         // scan chunk length
#define NC 16          // ceil(L/CHK) == ceil((L-1)/CHK)

typedef __attribute__((ext_vector_type(16))) __bf16 bf16x16;
typedef __attribute__((ext_vector_type(8)))  __bf16 bf16x8;
typedef __attribute__((ext_vector_type(4)))  __bf16 bf16x4;
typedef __attribute__((ext_vector_type(8)))  float  f32x8;
typedef unsigned int u32x4 __attribute__((ext_vector_type(4)));
typedef int          i32x8 __attribute__((ext_vector_type(8)));
typedef int          i32x4 __attribute__((ext_vector_type(4)));

#if defined(__has_builtin)
#if __has_builtin(__builtin_amdgcn_tensor_load_to_lds) && \
    __has_builtin(__builtin_amdgcn_s_wait_tensorcnt)
#define HAVE_TDM 1
#endif
#endif

static __device__ __forceinline__ __bf16 tobf(float f) { return (__bf16)f; }

static __device__ __forceinline__ f32x8 zero8() {
    f32x8 z;
#pragma unroll
    for (int i = 0; i < 8; ++i) z[i] = 0.0f;
    return z;
}

static __device__ __forceinline__ bf16x4 pack4(float4 v) {
    bf16x4 r;
    r[0] = tobf(v.x); r[1] = tobf(v.y); r[2] = tobf(v.z); r[3] = tobf(v.w);
    return r;
}

// 16-bit WMMA fragment from a row-major LDS tile: per-lane row = m0 + (lane&15),
// registers walk K; each lane holds two 8-element K runs (k0+8h, k0+16+8h) so this is exactly two
// 16-byte ds_load_b128 per lane (stride must be a multiple of 8 elements).
static __device__ __forceinline__ bf16x16 load_row(const __bf16* lds, int stride,
                                                   int m0, int k0, int lane) {
    const int half = (lane >> 4) & 1;
    const __bf16* p = lds + (m0 + (lane & 15)) * stride + k0 + half * 8;
    bf16x8 lo = *(const bf16x8*)(p);
    bf16x8 hi = *(const bf16x8*)(p + 16);
    bf16x16 f;
#pragma unroll
    for (int i = 0; i < 8; ++i) {
        f[i]     = lo[i];
        f[i + 8] = hi[i];
    }
    return f;
}

static __device__ __forceinline__ f32x8 wmma_bf16(bf16x16 a, bf16x16 b, f32x8 c) {
    return __builtin_amdgcn_wmma_f32_16x16x32_bf16(false, a, false, b, (short)0, c,
                                                   false, false);
}

// ---------------------------------------------------------------------------
// Pre-conversion kernels: fp32 -> bf16 once, so GEMMs read bf16 (half HBM
// traffic; x is consumed by 4 GEMMs) and tiles become plain 2-D bf16 copies.
// ---------------------------------------------------------------------------
__global__ void tobf16_kernel(const float* __restrict__ src, __bf16* __restrict__ dst,
                              long n4) {
    for (long i4 = (long)blockIdx.x * blockDim.x + threadIdx.x; i4 < n4;
         i4 += (long)gridDim.x * blockDim.x) {
        float4 v = *(const float4*)(src + i4 * 4);
        *(bf16x4*)(dst + i4 * 4) = pack4(v);
    }
}

// Wt[n][k] = (bf16) W[k][n] : 64x64 tile transpose through LDS.
__global__ __launch_bounds__(256) void wtranspose_kernel(const float* __restrict__ W,
                                                         __bf16* __restrict__ Wt) {
    __shared__ __align__(16) __bf16 Ts[64 * 72];  // [k][n]
    const int tid = threadIdx.x;
    const int n0 = blockIdx.x * 64, k0 = blockIdx.y * 64;
#pragma unroll
    for (int i = 0; i < 4; ++i) {
        int idx4 = i * 256 + tid;  // 1024 segments of 4
        int r = idx4 >> 4, cs = (idx4 & 15) * 4;
        float4 v = *(const float4*)(W + (long)(k0 + r) * DD + n0 + cs);
        *(bf16x4*)(Ts + r * 72 + cs) = pack4(v);
    }
    __syncthreads();
    const int r2 = tid & 63;         // output row (n)
    const int ks = (tid >> 6) * 16;  // k segment
    bf16x8 p0, p1;
#pragma unroll
    for (int j = 0; j < 8; ++j) {
        p0[j] = Ts[(ks + j) * 72 + r2];
        p1[j] = Ts[(ks + 8 + j) * 72 + r2];
    }
    *(bf16x8*)(Wt + (long)(n0 + r2) * DD + k0 + ks)     = p0;
    *(bf16x8*)(Wt + (long)(n0 + r2) * DD + k0 + ks + 8) = p1;
}

// ---------------------------------------------------------------------------
// GEMM with bias: Out[M,N] = A[M,K] @ W[K,N] + bias[N]
// A is bf16 row-major [M][K]; W is pre-transposed bf16 Bt[N][K].
// Staging via Tensor Data Mover (double-buffered), or wide bf16 loads fallback.
// TDM pad: 16 dwords (64B row) + 4 dwords (16B) pad == 80B pitch == GSTR bf16.
// ---------------------------------------------------------------------------
#define GBM 128
#define GBN 128
#define GBK 32
#define GSTR 40  // 32 + 8 pad elements (80B pitch)

#if HAVE_TDM
static __device__ __forceinline__ void tdm_load_tile(const __bf16* gbase,
                                                     unsigned ldsOff,
                                                     int tileRows, int tensorRows) {
    unsigned long long ga = (unsigned long long)(uintptr_t)gbase;
    u32x4 g0;
    g0[0] = 1u;                     // count=1, user descriptor
    g0[1] = ldsOff;                 // lds_addr (bytes)
    g0[2] = (unsigned)ga;           // global_addr[31:0]
    g0[3] = (unsigned)((ga >> 32) & 0x01FFFFFFu) | (2u << 30);  // addr[56:32]|type=2
    i32x8 g1;
    // data_size=2B (code 1), pad_enable, pad_interval=3 (16 dw), pad_amount=3 (4 dw)
    g1[0] = (1 << 16) | (1 << 20) | (3 << 22) | (3 << 25);
    g1[1] = (int)((DD & 0xFFFF) << 16);                  // tensor_dim0 = K (lo16)
    g1[2] = (int)((unsigned)(tensorRows & 0xFFFF) << 16);// dim0 hi=0, tensor_dim1 lo
    g1[3] = (32 << 16);                                  // dim1 hi=0, tile_dim0 = 32
    g1[4] = tileRows & 0xFFFF;                           // tile_dim1 ; tile_dim2=0
    g1[5] = DD;                                          // tensor_dim0_stride = K
    g1[6] = 0;
    g1[7] = 0;
    i32x4 z4;
    z4[0] = z4[1] = z4[2] = z4[3] = 0;
    i32x8 z8;
#pragma unroll
    for (int i = 0; i < 8; ++i) z8[i] = 0;
    // 6-arg form on this toolchain: (g0, g1, g2, g3, g4, cpol)
    __builtin_amdgcn_tensor_load_to_lds(g0, g1, z4, z4, z8, 0);
}
#endif

static __device__ __forceinline__ void gemm_tile_compute(const __bf16* As_,
                                                         const __bf16* Bts_, int wm,
                                                         int wn, int lane,
                                                         f32x8 (&acc)[2][4]) {
    bf16x16 af[2];
#pragma unroll
    for (int mi = 0; mi < 2; ++mi)
        af[mi] = load_row(As_, GSTR, wm * 32 + mi * 16, 0, lane);
#pragma unroll
    for (int ni = 0; ni < 4; ++ni) {
        bf16x16 bfr = load_row(Bts_, GSTR, wn * 64 + ni * 16, 0, lane);
#pragma unroll
        for (int mi = 0; mi < 2; ++mi)
            acc[mi][ni] = wmma_bf16(af[mi], bfr, acc[mi][ni]);
    }
}

__global__ __launch_bounds__(256) void gemm_bias_bf16_kernel(
    const __bf16* __restrict__ A, const __bf16* __restrict__ Bt,
    const float* __restrict__ bias, float* __restrict__ Out,
    int M, int N, int K) {
    __shared__ __align__(16) __bf16 As[2][GBM * GSTR];   // [m][k], double buffer
    __shared__ __align__(16) __bf16 Bts[2][GBN * GSTR];  // [n][k], double buffer

    const int tid  = threadIdx.x;
    const int lane = tid & 31;
    const int wave = tid >> 5;
    const int wm   = wave >> 1;  // 0..3
    const int wn   = wave & 1;   // 0..1
    const int row0 = blockIdx.y * GBM;
    const int col0 = blockIdx.x * GBN;

    f32x8 acc[2][4];
#pragma unroll
    for (int i = 0; i < 2; ++i)
#pragma unroll
        for (int j = 0; j < 4; ++j) acc[i][j] = zero8();

#if HAVE_TDM
    if (wave == 0) {
        tdm_load_tile(A + (long)row0 * K, (unsigned)(uintptr_t)&As[0][0], GBM, M);
        tdm_load_tile(Bt + (long)col0 * K, (unsigned)(uintptr_t)&Bts[0][0], GBN, N);
        __builtin_amdgcn_s_wait_tensorcnt(0);
    }
    __syncthreads();
    int buf = 0;
    for (int kt = 0; kt < K; kt += GBK, buf ^= 1) {
        if (wave == 0 && kt + GBK < K) {
            // issue next tiles into the other buffer; overlaps this compute
            tdm_load_tile(A + (long)row0 * K + kt + GBK,
                          (unsigned)(uintptr_t)&As[buf ^ 1][0], GBM, M);
            tdm_load_tile(Bt + (long)col0 * K + kt + GBK,
                          (unsigned)(uintptr_t)&Bts[buf ^ 1][0], GBN, N);
        }
        gemm_tile_compute(&As[buf][0], &Bts[buf][0], wm, wn, lane, acc);
        __syncthreads();
        if (wave == 0) __builtin_amdgcn_s_wait_tensorcnt(0);
        __syncthreads();
    }
#else
    for (int kt = 0; kt < K; kt += GBK) {
        bf16x8 ta[2], tb[2];
#pragma unroll
        for (int i = 0; i < 2; ++i) {
            int idx8 = i * 256 + tid;  // 512 segments of 8
            int r = idx8 >> 2, cs = (idx8 & 3) * 8;
            ta[i] = *(const bf16x8*)(A + (long)(row0 + r) * K + kt + cs);
            tb[i] = *(const bf16x8*)(Bt + (long)(col0 + r) * K + kt + cs);
        }
        if (kt + GBK < K) {
            __builtin_prefetch(&A[(long)(row0 + (tid >> 1)) * K + kt + GBK], 0, 1);
            __builtin_prefetch(&Bt[(long)(col0 + (tid >> 1)) * K + kt + GBK], 0, 1);
        }
        __syncthreads();
#pragma unroll
        for (int i = 0; i < 2; ++i) {
            int idx8 = i * 256 + tid;
            int r = idx8 >> 2, cs = (idx8 & 3) * 8;
            *(bf16x8*)(&As[0][0] + r * GSTR + cs)  = ta[i];
            *(bf16x8*)(&Bts[0][0] + r * GSTR + cs) = tb[i];
        }
        __syncthreads();
        gemm_tile_compute(&As[0][0], &Bts[0][0], wm, wn, lane, acc);
    }
    __syncthreads();
#endif

    const int half = lane >> 4;
#pragma unroll
    for (int mi = 0; mi < 2; ++mi)
#pragma unroll
        for (int ni = 0; ni < 4; ++ni) {
            int m0 = row0 + wm * 32 + mi * 16;
            int n  = col0 + wn * 64 + ni * 16 + (lane & 15);
            float bv = bias[n];
#pragma unroll
            for (int v = 0; v < 8; ++v) {
                int m = m0 + v + half * 8;
                Out[(long)m * N + n] = acc[mi][ni][v] + bv;
            }
        }
}

// ---------------------------------------------------------------------------
// Chunked linear-attention scan kernels. One block = one (b, h, chunk).
// Buffers are (B, L, H*Dh) fp32; head h occupies cols [h*64, h*64+64).
// ---------------------------------------------------------------------------
#define CSTR 72  // 64 + 8 pad, multiple of 8 (16B-aligned rows)

// Pass A: G_c[d][e] = sum_{t in chunk} K[t][d] * V[t][e]   (= K_c^T V_c)
__global__ __launch_bounds__(128) void chunk_kv_kernel(
    const float* __restrict__ Kp, const float* __restrict__ Vp,
    float* __restrict__ G, int seqLen) {
    __shared__ __align__(16) __bf16 Kts[DH * CSTR];  // [d][t]
    __shared__ __align__(16) __bf16 Vts[DH * CSTR];  // [e][t]

    const int tid = threadIdx.x, lane = tid & 31, wave = tid >> 5;
    const int blk = blockIdx.x;
    const int c = blk % NC;
    const int h = (blk / NC) % HH;
    const int b = blk / (NC * HH);
    const int t0 = c * CHK;

    {
        const int d = tid & 63, tseg = (tid >> 6) * 32;
        float kreg[32], vreg[32];
#pragma unroll
        for (int j = 0; j < 32; ++j) {
            int gt = t0 + tseg + j;
            float kv = 0.0f, vv = 0.0f;
            if (gt < seqLen) {
                long off = ((long)(b * LL + gt)) * DD + h * DH + d;
                kv = Kp[off];
                vv = Vp[off];
            }
            kreg[j] = kv;
            vreg[j] = vv;
        }
#pragma unroll
        for (int s = 0; s < 4; ++s) {
            bf16x8 pk, pv;
#pragma unroll
            for (int j = 0; j < 8; ++j) { pk[j] = tobf(kreg[s * 8 + j]); pv[j] = tobf(vreg[s * 8 + j]); }
            *(bf16x8*)(Kts + d * CSTR + tseg + s * 8) = pk;
            *(bf16x8*)(Vts + d * CSTR + tseg + s * 8) = pv;
        }
    }
    __syncthreads();

    const int d0 = wave * 16;
    f32x8 acc[4];
#pragma unroll
    for (int j = 0; j < 4; ++j) acc[j] = zero8();

#pragma unroll
    for (int kk = 0; kk < CHK; kk += 32) {
        bf16x16 af = load_row(Kts, CSTR, d0, kk, lane);
#pragma unroll
        for (int ni = 0; ni < 4; ++ni) {
            bf16x16 bfr = load_row(Vts, CSTR, ni * 16, kk, lane);
            acc[ni] = wmma_bf16(af, bfr, acc[ni]);
        }
    }

    float* gout = G + ((long)((b * HH + h) * NC + c)) * (DH * DH);
    const int half = lane >> 4;
#pragma unroll
    for (int ni = 0; ni < 4; ++ni) {
        int e = ni * 16 + (lane & 15);
#pragma unroll
        for (int v = 0; v < 8; ++v) {
            int d = d0 + v + half * 8;
            gout[d * DH + e] = acc[ni][v];
        }
    }
}

// Pass B: in-place exclusive prefix over chunk states. One block per (b,h).
__global__ __launch_bounds__(256) void prefix_kernel(float* __restrict__ G, int nchunks) {
    float* base = G + (long)blockIdx.x * nchunks * (DH * DH);
    for (int e = threadIdx.x; e < DH * DH; e += blockDim.x) {
        float run = 0.0f;
        for (int c = 0; c < nchunks; ++c) {
            float g = base[c * (DH * DH) + e];
            base[c * (DH * DH) + e] = run;
            run += g;
        }
    }
}

// Pass C: O_c = tril(Q_c K_c^T) V_c + Q_c S_c   (inclusive causal mask)
__global__ __launch_bounds__(128) void chunk_out_kernel(
    const float* __restrict__ Qp, const float* __restrict__ Kp,
    const float* __restrict__ Vp, const float* __restrict__ S,
    float* __restrict__ Op, int seqLen) {
    __shared__ __align__(16) __bf16 Qs[CHK * CSTR];   // [t][d]
    __shared__ __align__(16) __bf16 Ks[CHK * CSTR];   // [t'][d]
    __shared__ __align__(16) __bf16 Vts[DH * CSTR];   // [e][t']
    __shared__ __align__(16) __bf16 Sts[DH * CSTR];   // [e][d]
    __shared__ __align__(16) __bf16 Ps[CHK * CSTR];   // [t][t']

    const int tid = threadIdx.x, lane = tid & 31, wave = tid >> 5;
    const int blk = blockIdx.x;
    const int c = blk % NC;
    const int h = (blk / NC) % HH;
    const int b = blk / (NC * HH);
    const int t0 = c * CHK;

    const float* Sg = S + ((long)((b * HH + h) * NC + c)) * (DH * DH);

#pragma unroll
    for (int i = 0; i < 8; ++i) {
        int idx4 = i * 128 + tid;
        int t = idx4 >> 4, cs = (idx4 & 15) * 4;
        int gt = t0 + t;
        float4 q4 = make_float4(0.f, 0.f, 0.f, 0.f);
        float4 k4 = make_float4(0.f, 0.f, 0.f, 0.f);
        if (gt < seqLen) {
            long off = ((long)(b * LL + gt)) * DD + h * DH + cs;
            q4 = *(const float4*)(Qp + off);
            k4 = *(const float4*)(Kp + off);
        }
        *(bf16x4*)(Qs + t * CSTR + cs) = pack4(q4);
        *(bf16x4*)(Ks + t * CSTR + cs) = pack4(k4);
    }
    {
        const int e = tid & 63, rseg = (tid >> 6) * 32;
        float vreg[32], sreg[32];
#pragma unroll
        for (int j = 0; j < 32; ++j) {
            int r = rseg + j;
            int gt = t0 + r;
            float vv = 0.0f;
            if (gt < seqLen)
                vv = Vp[((long)(b * LL + gt)) * DD + h * DH + e];
            vreg[j] = vv;
            sreg[j] = Sg[r * DH + e];
        }
#pragma unroll
        for (int s = 0; s < 4; ++s) {
            bf16x8 pv, ps;
#pragma unroll
            for (int j = 0; j < 8; ++j) { pv[j] = tobf(vreg[s * 8 + j]); ps[j] = tobf(sreg[s * 8 + j]); }
            *(bf16x8*)(Vts + e * CSTR + rseg + s * 8) = pv;
            *(bf16x8*)(Sts + e * CSTR + rseg + s * 8) = ps;
        }
    }
    __syncthreads();

    const int m0 = wave * 16;
    const int half = lane >> 4;

    // Phase 1: P = Q K^T with causal (inclusive) mask
    f32x8 pacc[4];
#pragma unroll
    for (int j = 0; j < 4; ++j) pacc[j] = zero8();
#pragma unroll
    for (int kk = 0; kk < DH; kk += 32) {
        bf16x16 af = load_row(Qs, CSTR, m0, kk, lane);
#pragma unroll
        for (int ni = 0; ni < 4; ++ni) {
            bf16x16 bfr = load_row(Ks, CSTR, ni * 16, kk, lane);
            pacc[ni] = wmma_bf16(af, bfr, pacc[ni]);
        }
    }
#pragma unroll
    for (int ni = 0; ni < 4; ++ni) {
        int col = ni * 16 + (lane & 15);
#pragma unroll
        for (int v = 0; v < 8; ++v) {
            int row = m0 + v + half * 8;
            float pv = (col <= row) ? pacc[ni][v] : 0.0f;
            Ps[row * CSTR + col] = tobf(pv);
        }
    }
    __syncthreads();

    // Phase 2: O = P @ V + Q @ S
    f32x8 oacc[4];
#pragma unroll
    for (int j = 0; j < 4; ++j) oacc[j] = zero8();
#pragma unroll
    for (int kk = 0; kk < CHK; kk += 32) {
        bf16x16 afp = load_row(Ps, CSTR, m0, kk, lane);
        bf16x16 afq = load_row(Qs, CSTR, m0, kk, lane);
#pragma unroll
        for (int ni = 0; ni < 4; ++ni) {
            bf16x16 bfv = load_row(Vts, CSTR, ni * 16, kk, lane);
            oacc[ni] = wmma_bf16(afp, bfv, oacc[ni]);
            bf16x16 bfs = load_row(Sts, CSTR, ni * 16, kk, lane);
            oacc[ni] = wmma_bf16(afq, bfs, oacc[ni]);
        }
    }
#pragma unroll
    for (int ni = 0; ni < 4; ++ni) {
        int e = ni * 16 + (lane & 15);
#pragma unroll
        for (int v = 0; v < 8; ++v) {
            int t = m0 + v + half * 8;
            int gt = t0 + t;
            if (gt < seqLen)
                Op[((long)(b * LL + gt)) * DD + h * DH + e] = oacc[ni][v];
        }
    }
}

// ---------------------------------------------------------------------------
// Elementwise kernels (float4-vectorized)
// ---------------------------------------------------------------------------
__global__ void transform_kernel(float* __restrict__ Qb, float* __restrict__ K2b,
                                 float* __restrict__ Eb, const float* __restrict__ x,
                                 const float* __restrict__ O1) {
    const long total4 = (long)BB * LL * DD / 4;
    for (long i4 = (long)blockIdx.x * blockDim.x + threadIdx.x; i4 < total4;
         i4 += (long)gridDim.x * blockDim.x) {
        long i = i4 * 4;
        float4 q = *(const float4*)(Qb + i);
        float4 k = *(const float4*)(K2b + i);
        float* qa = &q.x;
        float* ka = &k.x;
#pragma unroll
        for (int j = 0; j < 4; ++j) {
            float a = qa[j] * 0.125f;           // 1/sqrt(Dh)
            qa[j] = (a <= 0.0f) ? a : 0.02f * a;
            float z = ka[j] * (0.02f / 32.0f);  // 0.02 / sqrt(D)
            ka[j] = 1.0f / (1.0f + __expf(-z));
        }
        *(float4*)(Qb + i) = q;
        *(float4*)(K2b + i) = k;

        long rem = i % ((long)LL * DD);
        int t = (int)(rem / DD);
        if (t < LL - 1) {
            float4 xv = *(const float4*)(x + i + DD);
            float4 ov = *(const float4*)(O1 + i);
            *(float4*)(Eb + i) =
                make_float4(xv.x - ov.x, xv.y - ov.y, xv.z - ov.z, xv.w - ov.w);
        } else {
            *(float4*)(Eb + i) = make_float4(0.f, 0.f, 0.f, 0.f);
        }
    }
}

// yb (bf16) = O1 + shift_right(O2) : feeds the final bf16 GEMM directly.
__global__ void combine_kernel(const float* __restrict__ O1,
                               const float* __restrict__ O2,
                               __bf16* __restrict__ Yb) {
    const long total4 = (long)BB * LL * DD / 4;
    for (long i4 = (long)blockIdx.x * blockDim.x + threadIdx.x; i4 < total4;
         i4 += (long)gridDim.x * blockDim.x) {
        long i = i4 * 4;
        long rem = i % ((long)LL * DD);
        int t = (int)(rem / DD);
        float4 y = *(const float4*)(O1 + i);
        if (t > 0) {
            float4 o = *(const float4*)(O2 + i - DD);
            y.x += o.x; y.y += o.y; y.z += o.z; y.w += o.w;
        }
        *(bf16x4*)(Yb + i) = pack4(y);
    }
}

// ---------------------------------------------------------------------------
extern "C" void kernel_launch(void* const* d_in, const int* in_sizes, int n_in,
                              void* d_out, int out_size, void* d_ws, size_t ws_size,
                              hipStream_t stream) {
    (void)in_sizes; (void)n_in; (void)out_size; (void)ws_size;

    const float* x   = (const float*)d_in[0];
    const float* Wq  = (const float*)d_in[1];
    const float* bq  = (const float*)d_in[2];
    const float* Wk1 = (const float*)d_in[3];
    const float* bk1 = (const float*)d_in[4];
    const float* Wv  = (const float*)d_in[5];
    const float* bv  = (const float*)d_in[6];
    const float* Wk2 = (const float*)d_in[7];
    const float* bk2 = (const float*)d_in[8];
    const float* Wp  = (const float*)d_in[9];
    const float* bp  = (const float*)d_in[10];
    float* out = (float*)d_out;

    const long NBLD = (long)BB * LL * DD;  // 2,097,152 floats per fp32 buffer
    float* ws  = (float*)d_ws;
    float* Qb  = ws + 0 * NBLD;  // Q  -> q2 (in place)
    float* Kb  = ws + 1 * NBLD;  // K1 -> E  (in place after branch 1)
    float* Vb  = ws + 2 * NBLD;  // V  -> O2 (reused after branch 1)
    float* K2b = ws + 3 * NBLD;  // K2raw -> k2 (in place)
    float* O1b = ws + 4 * NBLD;  // O1
    float* Gb  = ws + 5 * NBLD;  // chunk states
    float* O2b = Vb;

    // bf16 pool after the fp32 buffers
    __bf16* bfp = (__bf16*)(ws + 6 * NBLD);
    __bf16* xb  = bfp;               // [M][K]
    __bf16* yb  = bfp + NBLD;        // [M][K]
    __bf16* Wtp5 = bfp + 2 * NBLD;   // 5 transposed weights, each DD*DD
    __bf16* WtQ  = Wtp5 + 0L * DD * DD;
    __bf16* WtK1 = Wtp5 + 1L * DD * DD;
    __bf16* WtV  = Wtp5 + 2L * DD * DD;
    __bf16* WtK2 = Wtp5 + 3L * DD * DD;
    __bf16* WtP  = Wtp5 + 4L * DD * DD;

    // Pre-convert: x -> bf16 ; W* -> transposed bf16
    tobf16_kernel<<<1024, 256, 0, stream>>>(x, xb, NBLD / 4);
    dim3 tg(DD / 64, DD / 64);  // (16,16)
    wtranspose_kernel<<<tg, 256, 0, stream>>>(Wq,  WtQ);
    wtranspose_kernel<<<tg, 256, 0, stream>>>(Wk1, WtK1);
    wtranspose_kernel<<<tg, 256, 0, stream>>>(Wv,  WtV);
    wtranspose_kernel<<<tg, 256, 0, stream>>>(Wk2, WtK2);
    wtranspose_kernel<<<tg, 256, 0, stream>>>(Wp,  WtP);

    dim3 gg(DD / GBN, MM / GBM);  // (8, 16)
    gemm_bias_bf16_kernel<<<gg, 256, 0, stream>>>(xb, WtQ,  bq,  Qb,  MM, DD, DD);
    gemm_bias_bf16_kernel<<<gg, 256, 0, stream>>>(xb, WtK1, bk1, Kb,  MM, DD, DD);
    gemm_bias_bf16_kernel<<<gg, 256, 0, stream>>>(xb, WtV,  bv,  Vb,  MM, DD, DD);
    gemm_bias_bf16_kernel<<<gg, 256, 0, stream>>>(xb, WtK2, bk2, K2b, MM, DD, DD);

    const int nblk = BB * HH * NC;  // 512
    // AR branch
    chunk_kv_kernel<<<nblk, 128, 0, stream>>>(Kb, Vb, Gb, LL);
    prefix_kernel<<<BB * HH, 256, 0, stream>>>(Gb, NC);
    chunk_out_kernel<<<nblk, 128, 0, stream>>>(Qb, Kb, Vb, Gb, O1b, LL);

    // Build MA-branch operands: q2 (in Qb), k2 (in K2b), E (in Kb)
    transform_kernel<<<1024, 256, 0, stream>>>(Qb, K2b, Kb, x, O1b);

    // MA branch over L-1 steps (tail zero-filled inside chunk kernels)
    chunk_kv_kernel<<<nblk, 128, 0, stream>>>(K2b, Kb, Gb, LL - 1);
    prefix_kernel<<<BB * HH, 256, 0, stream>>>(Gb, NC);
    chunk_out_kernel<<<nblk, 128, 0, stream>>>(Qb, K2b, Kb, Gb, O2b, LL - 1);

    // y (bf16) = O1 + shift(O2), then final projection
    combine_kernel<<<1024, 256, 0, stream>>>(O1b, O2b, yb);
    gemm_bias_bf16_kernel<<<gg, 256, 0, stream>>>(yb, WtP, bp, out, MM, DD, DD);
}
